// IntersiteModel_60859686585094
// MI455X (gfx1250) — compile-verified, matrix operand order checked by reading, not run on hardware
//
#include <hip/hip_runtime.h>
#include <hip/hip_bf16.h>
#include <math.h>

typedef __attribute__((ext_vector_type(16))) __bf16 v16bf;
typedef __attribute__((ext_vector_type(8)))  __bf16 v8bf;
typedef __attribute__((ext_vector_type(8)))  float  v8f;

// ---------------- workspace layout (bf16 elements) ----------------
#define WS_S1L0 0u        // K=1280, M=64
#define WS_S1L1 81920u    // K=1280, M=32
#define WS_S2L0 122880u   // K= 768, M=96
#define WS_S2L1 196608u   // K=1024, M=32
#define WS_S3L0 229376u   // K=5120, M=96
#define WS_S3L1 720896u   // K=5120, M=32
#define WS_S4   884736u   // K=5120, M=16 (col0 live, rest zero)
#define WS_TOTAL 966656u

// ---------------- weight prep: fp32 w[u,v,w] -> bf16 WMMA B fragments ----
// Fragment order: [kstep][ntile][lane(32)][i(16)], lane = (col&15)+16*kg,
// K(kg,i) = 8*kg + (i&7) + ((i>>3)<<4)   (16-bit operand layout, ISA 7.12.2)
__global__ void prep_w(const float* __restrict__ w, __bf16* __restrict__ Bp,
                       int total, int ncols, int Koff, int m2, int m3,
                       int coff, int Mpad, float scale)
{
  int e = blockIdx.x * blockDim.x + threadIdx.x;
  if (e >= total) return;
  int kl = e / ncols, c = e - kl * ncols;
  int u  = kl / m2,   v = kl - u * m2;
  float val = w[(u * m2 + v) * m3 + c] * scale;
  int k  = Koff + kl;
  int ks = k >> 5, kr = k & 31;
  int kg = (kr >> 3) & 1;
  int i  = (kr & 7) + ((kr >> 4) << 3);
  int col = coff + c;
  int lane = (col & 15) + (kg << 4);
  int nt = col >> 4;
  Bp[(((size_t)ks * (Mpad >> 4) + nt) * 32 + lane) * 16 + i] = (__bf16)val;
}

// ---------------- device helpers ----------------
__device__ __forceinline__ float bf(__bf16 x) { return (float)x; }

struct alignas(16) R8  { __bf16 s[8];  };
struct alignas(16) R24 { __bf16 s[24]; };
__device__ __forceinline__ R8 ld8(const __bf16* p) {
  R8 r; *(v8bf*)r.s = *(const v8bf*)p; return r;
}
__device__ __forceinline__ R24 ld24(const __bf16* p) {
  R24 r;
  *(v8bf*)(r.s)      = *(const v8bf*)p;
  *(v8bf*)(r.s + 8)  = *(const v8bf*)(p + 8);
  *(v8bf*)(r.s + 16) = *(const v8bf*)(p + 16);
  return r;
}

template<int NT> struct BB {
  v16bf b[NT];
  __device__ __forceinline__ void load(const v16bf* __restrict__ Bv, int ks, int lane) {
#pragma unroll
    for (int t = 0; t < NT; ++t) b[t] = Bv[(ks * NT + t) * 32 + lane];
  }
  __device__ __forceinline__ void fire(const v16bf& a, v8f* acc) {
#pragma unroll
    for (int t = 0; t < NT; ++t)
      acc[t] = __builtin_amdgcn_wmma_f32_16x16x32_bf16(
          false, a, false, b[t], (short)0, acc[t], false, false);
  }
};

// ---- stage1 L1 pass (vector outputs), KC = component ----
template<int KC>
__device__ __forceinline__ void s1l1(const __bf16* __restrict__ Bp,
                                     const __bf16* x1r, const __bf16* x2r,
                                     __bf16* H, int lane, int row, int kg)
{
  constexpr int C1 = (KC + 1) % 3, C2 = (KC + 2) % 3;
  v8f acc[2] = {};
  const v16bf* Bv = (const v16bf*)Bp;
  BB<2> bb;
  R24 yv = ld24(x2r + 32 + 24 * kg);          // X2V(v=8kg+q, c), shared by segA/C
  for (int ks = 0; ks < 16; ++ks) {           // segA: s1(u)*v2(v,KC), u=2ks(+1)
    bb.load(Bv, ks, lane);
    float x0 = bf(x1r[2 * ks]), x1_ = bf(x1r[2 * ks + 1]);
    v16bf a;
#pragma unroll
    for (int q = 0; q < 8; ++q) { float yq = bf(yv.s[3 * q + KC]);
      a[q] = (__bf16)(x0 * yq); a[q + 8] = (__bf16)(x1_ * yq); }
    bb.fire(a, acc);
  }
  R8 plo = ld8(x2r + 8 * kg), phi = ld8(x2r + 16 + 8 * kg);
  for (int ks = 16; ks < 32; ++ks) {          // segB: v1(u,KC)*s2(v), u=ks-16
    bb.load(Bv, ks, lane);
    float xu = bf(x1r[32 + 3 * (ks - 16) + KC]);
    v16bf a;
#pragma unroll
    for (int q = 0; q < 8; ++q) {
      a[q] = (__bf16)(xu * bf(plo.s[q])); a[q + 8] = (__bf16)(xu * bf(phi.s[q])); }
    bb.fire(a, acc);
  }
  for (int ks = 32; ks < 40; ++ks) {          // segC: cross(v1,v2)[KC]
    bb.load(Bv, ks, lane);
    const __bf16* xa = x1r + 32 + 6 * (ks - 32);
    float u0c1 = bf(xa[C1]), u0c2 = bf(xa[C2]);
    float u1c1 = bf(xa[3 + C1]), u1c2 = bf(xa[3 + C2]);
    v16bf a;
#pragma unroll
    for (int q = 0; q < 8; ++q) {
      float yc1 = bf(yv.s[3 * q + C1]), yc2 = bf(yv.s[3 * q + C2]);
      a[q]     = (__bf16)(u0c1 * yc2 - u0c2 * yc1);
      a[q + 8] = (__bf16)(u1c1 * yc2 - u1c2 * yc1);
    }
    bb.fire(a, acc);
  }
#pragma unroll
  for (int t = 0; t < 2; ++t)
#pragma unroll
    for (int rr = 0; rr < 8; ++rr)
      H[(rr + 8 * kg) * 160 + 64 + (t * 16 + row) * 3 + KC] = (__bf16)acc[t][rr];
}

// ---- stage2 L1 pass ----
template<int KC>
__device__ __forceinline__ void s2l1(const __bf16* __restrict__ Bp,
                                     const __bf16* hr, const __bf16* er,
                                     float* T, int lane, int row, int kg)
{
  constexpr int C1 = (KC + 1) % 3, C2 = (KC + 2) % 3;
  v8f acc[2] = {};
  const v16bf* Bv = (const v16bf*)Bp;
  BB<2> bb;
  R24 ev = ld24(er + 8);                       // EV(v=0..7, c)
  for (int ks = 0; ks < 16; ++ks) {            // segA: hs(u)*ev(v,KC)
    bb.load(Bv, ks, lane);
    float x0 = bf(hr[4 * ks + kg]), x1_ = bf(hr[4 * ks + 2 + kg]);
    v16bf a;
#pragma unroll
    for (int q = 0; q < 8; ++q) { float eq = bf(ev.s[3 * q + KC]);
      a[q] = (__bf16)(x0 * eq); a[q + 8] = (__bf16)(x1_ * eq); }
    bb.fire(a, acc);
  }
  R8 es = ld8(er);
  for (int ks = 16; ks < 24; ++ks) {           // segB: hv(u,KC)*es(v)
    bb.load(Bv, ks, lane);
    int u0 = 4 * (ks - 16) + kg;
    float x0 = bf(hr[64 + 3 * u0 + KC]), x1_ = bf(hr[64 + 3 * (u0 + 2) + KC]);
    v16bf a;
#pragma unroll
    for (int q = 0; q < 8; ++q) { float eq = bf(es.s[q]);
      a[q] = (__bf16)(x0 * eq); a[q + 8] = (__bf16)(x1_ * eq); }
    bb.fire(a, acc);
  }
  for (int ks = 24; ks < 32; ++ks) {           // segC: cross(hv,ev)[KC]
    bb.load(Bv, ks, lane);
    int u0 = 4 * (ks - 24) + kg;
    const __bf16* h0 = hr + 64 + 3 * u0;
    float xc1 = bf(h0[C1]), xc2 = bf(h0[C2]);
    float yc1 = bf(h0[6 + C1]), yc2 = bf(h0[6 + C2]);
    v16bf a;
#pragma unroll
    for (int q = 0; q < 8; ++q) {
      float ec1 = bf(ev.s[3 * q + C1]), ec2 = bf(ev.s[3 * q + C2]);
      a[q]     = (__bf16)(xc1 * ec2 - xc2 * ec1);
      a[q + 8] = (__bf16)(yc1 * ec2 - yc2 * ec1);
    }
    bb.fire(a, acc);
  }
#pragma unroll
  for (int t = 0; t < 2; ++t)
#pragma unroll
    for (int rr = 0; rr < 8; ++rr)
      T[(rr + 8 * kg) * 192 + 96 + (t * 16 + row) * 3 + KC] = acc[t][rr];
}

// ---- H x H -> l3=0 outputs (used by stage3 L0 with NT=6 and stage4 with NT=1)
template<int NT>
__device__ __forceinline__ void hh_l0(const __bf16* __restrict__ Bp,
                                      const __bf16* hr, v8f* acc, int lane, int kg)
{
  const v16bf* Bv = (const v16bf*)Bp;
  BB<NT> bb0, bb1;
  R8 v0l = ld8(hr + 8 * kg),      v0h = ld8(hr + 16 + 8 * kg);
  R8 v1l = ld8(hr + 32 + 8 * kg), v1h = ld8(hr + 48 + 8 * kg);
  for (int k2 = 0; k2 < 64; ++k2) {            // segA: hs(k2)*hs(v); ks=2k2,2k2+1
    bb0.load(Bv, 2 * k2, lane);
    bb1.load(Bv, 2 * k2 + 1, lane);
    float xu = bf(hr[k2]);
    v16bf a;
#pragma unroll
    for (int q = 0; q < 8; ++q) {
      a[q] = (__bf16)(xu * bf(v0l.s[q])); a[q + 8] = (__bf16)(xu * bf(v0h.s[q])); }
    bb0.fire(a, acc);
#pragma unroll
    for (int q = 0; q < 8; ++q) {
      a[q] = (__bf16)(xu * bf(v1l.s[q])); a[q + 8] = (__bf16)(xu * bf(v1h.s[q])); }
    bb1.fire(a, acc);
  }
  R24 wl = ld24(hr + 64 + 24 * kg), wh = ld24(hr + 112 + 24 * kg);
  for (int ks = 128; ks < 160; ++ks) {         // segB: dot(hv(u),hv(v)); u=ks-128
    bb0.load(Bv, ks, lane);
    const __bf16* h0 = hr + 64 + 3 * (ks - 128);
    float p0 = bf(h0[0]), p1 = bf(h0[1]), p2 = bf(h0[2]);
    v16bf a;
#pragma unroll
    for (int q = 0; q < 8; ++q) {
      a[q]     = (__bf16)(p0 * bf(wl.s[3*q]) + p1 * bf(wl.s[3*q+1]) + p2 * bf(wl.s[3*q+2]));
      a[q + 8] = (__bf16)(p0 * bf(wh.s[3*q]) + p1 * bf(wh.s[3*q+1]) + p2 * bf(wh.s[3*q+2]));
    }
    bb0.fire(a, acc);
  }
}

// ---- stage3 L1 pass ----
template<int KC>
__device__ __forceinline__ void s3l1(const __bf16* __restrict__ Bp,
                                     const __bf16* hr, float* T,
                                     int lane, int row, int kg)
{
  constexpr int C1 = (KC + 1) % 3, C2 = (KC + 2) % 3;
  v8f acc[2] = {};
  const v16bf* Bv = (const v16bf*)Bp;
  BB<2> bb0, bb1;
  R24 wl = ld24(hr + 64 + 24 * kg), wh = ld24(hr + 112 + 24 * kg);
  for (int ks = 0; ks < 64; ++ks) {            // segA: hs(ks)*hv(v,KC)
    bb0.load(Bv, ks, lane);
    float xu = bf(hr[ks]);
    v16bf a;
#pragma unroll
    for (int q = 0; q < 8; ++q) {
      a[q]     = (__bf16)(xu * bf(wl.s[3 * q + KC]));
      a[q + 8] = (__bf16)(xu * bf(wh.s[3 * q + KC]));
    }
    bb0.fire(a, acc);
  }
  R8 v0l = ld8(hr + 8 * kg),      v0h = ld8(hr + 16 + 8 * kg);
  R8 v1l = ld8(hr + 32 + 8 * kg), v1h = ld8(hr + 48 + 8 * kg);
  for (int k2 = 0; k2 < 32; ++k2) {            // segB: hv(k2,KC)*hs(v); ks=64+2k2(+1)
    bb0.load(Bv, 64 + 2 * k2, lane);
    bb1.load(Bv, 64 + 2 * k2 + 1, lane);
    float xu = bf(hr[64 + 3 * k2 + KC]);
    v16bf a;
#pragma unroll
    for (int q = 0; q < 8; ++q) {
      a[q] = (__bf16)(xu * bf(v0l.s[q])); a[q + 8] = (__bf16)(xu * bf(v0h.s[q])); }
    bb0.fire(a, acc);
#pragma unroll
    for (int q = 0; q < 8; ++q) {
      a[q] = (__bf16)(xu * bf(v1l.s[q])); a[q + 8] = (__bf16)(xu * bf(v1h.s[q])); }
    bb1.fire(a, acc);
  }
  for (int ks = 128; ks < 160; ++ks) {         // segC: cross(hv,hv)[KC]
    bb0.load(Bv, ks, lane);
    const __bf16* h0 = hr + 64 + 3 * (ks - 128);
    float xc1 = bf(h0[C1]), xc2 = bf(h0[C2]);
    v16bf a;
#pragma unroll
    for (int q = 0; q < 8; ++q) {
      a[q]     = (__bf16)(xc1 * bf(wl.s[3*q+C2]) - xc2 * bf(wl.s[3*q+C1]));
      a[q + 8] = (__bf16)(xc1 * bf(wh.s[3*q+C2]) - xc2 * bf(wh.s[3*q+C1]));
    }
    bb0.fire(a, acc);
  }
#pragma unroll
  for (int t = 0; t < 2; ++t)
#pragma unroll
    for (int rr = 0; rr < 8; ++rr)
      T[(rr + 8 * kg) * 192 + 96 + (t * 16 + row) * 3 + KC] = acc[t][rr];
}

__device__ __forceinline__ void apply_gate(const float* T, __bf16* H, int lane)
{
  for (int e = lane; e < 16 * 160; e += 32) {
    int r = e / 160, c = e - r * 160;
    float val;
    if (c < 64) { float s = T[r * 192 + 32 + c]; val = s / (1.f + __expf(-s)); }
    else { int j = c - 64; int u = j / 3;
      float g = T[r * 192 + u];
      val = (1.f / (1.f + __expf(-g))) * T[r * 192 + 96 + j]; }
    H[e] = (__bf16)val;
  }
}

// ---------------- fused end-to-end kernel: 1 wave = 16 rows ----------------
__global__ __launch_bounds__(64) void e3nn_fused(
    const float* __restrict__ gs1, const float* __restrict__ gs2,
    const float* __restrict__ ged, const __bf16* __restrict__ Bp,
    float* __restrict__ out)
{
  __shared__ __bf16 sX1[2][16 * 80];
  __shared__ __bf16 sX2[2][16 * 80];
  __shared__ __bf16 sE [2][16 * 32];
  __shared__ __bf16 sH [2][16 * 160];
  __shared__ float  sT [2][16 * 192];

  const int wave = threadIdx.x >> 5, lane = threadIdx.x & 31;
  const int row = lane & 15, kg = lane >> 4;
  __bf16 *X1 = sX1[wave], *X2 = sX2[wave], *E = sE[wave], *H = sH[wave];
  float  *T = sT[wave];
  const long base = ((long)blockIdx.x * 2 + wave) * 16;

  for (int e = lane; e < 16 * 80; e += 32) {
    X1[e] = (__bf16)gs1[base * 80 + e];
    X2[e] = (__bf16)gs2[base * 80 + e];
  }
  for (int e = lane; e < 16 * 32; e += 32) E[e] = (__bf16)ged[base * 32 + e];
  __syncthreads();

  const __bf16* x1r = X1 + row * 80;
  const __bf16* x2r = X2 + row * 80;
  const __bf16* er  = E  + row * 32;
  const __bf16* hr  = H  + row * 160;

  // ---- Stage 1: site1 x site2 -> H ----
  { v8f acc[4] = {};
    const v16bf* Bv = (const v16bf*)(Bp + WS_S1L0);
    BB<4> bb;
    R8 plo = ld8(x2r + 8 * kg), phi = ld8(x2r + 16 + 8 * kg);
    for (int ks = 0; ks < 32; ++ks) {          // segA: s1(ks)*s2(v)
      bb.load(Bv, ks, lane);
      float xu = bf(x1r[ks]);
      v16bf a;
#pragma unroll
      for (int q = 0; q < 8; ++q) {
        a[q] = (__bf16)(xu * bf(plo.s[q])); a[q + 8] = (__bf16)(xu * bf(phi.s[q])); }
      bb.fire(a, acc);
    }
    R24 yv = ld24(x2r + 32 + 24 * kg);
    for (int ks = 32; ks < 40; ++ks) {         // segB: dot(v1(u),v2(v))
      bb.load(Bv, ks, lane);
      const __bf16* xa = x1r + 32 + 6 * (ks - 32);
      float a0 = bf(xa[0]), a1 = bf(xa[1]), a2 = bf(xa[2]);
      float b0 = bf(xa[3]), b1 = bf(xa[4]), b2 = bf(xa[5]);
      v16bf a;
#pragma unroll
      for (int q = 0; q < 8; ++q) {
        a[q]     = (__bf16)(a0 * bf(yv.s[3*q]) + a1 * bf(yv.s[3*q+1]) + a2 * bf(yv.s[3*q+2]));
        a[q + 8] = (__bf16)(b0 * bf(yv.s[3*q]) + b1 * bf(yv.s[3*q+1]) + b2 * bf(yv.s[3*q+2]));
      }
      bb.fire(a, acc);
    }
#pragma unroll
    for (int t = 0; t < 4; ++t)
#pragma unroll
      for (int rr = 0; rr < 8; ++rr)
        H[(rr + 8 * kg) * 160 + t * 16 + row] = (__bf16)acc[t][rr];
  }
  s1l1<0>(Bp + WS_S1L1, x1r, x2r, H, lane, row, kg);
  s1l1<1>(Bp + WS_S1L1, x1r, x2r, H, lane, row, kg);
  s1l1<2>(Bp + WS_S1L1, x1r, x2r, H, lane, row, kg);
  __syncthreads();

  // ---- Stage 2: H x edge -> T ----
  { v8f acc[6] = {};
    const v16bf* Bv = (const v16bf*)(Bp + WS_S2L0);
    BB<6> bb;
    R8 es = ld8(er);
    for (int ks = 0; ks < 16; ++ks) {          // segA: hs(u)*es(v)
      bb.load(Bv, ks, lane);
      float x0 = bf(hr[4 * ks + kg]), x1_ = bf(hr[4 * ks + 2 + kg]);
      v16bf a;
#pragma unroll
      for (int q = 0; q < 8; ++q) { float eq = bf(es.s[q]);
        a[q] = (__bf16)(x0 * eq); a[q + 8] = (__bf16)(x1_ * eq); }
      bb.fire(a, acc);
    }
    R24 ev = ld24(er + 8);
    for (int ks = 16; ks < 24; ++ks) {         // segB: dot(hv(u),ev(v))
      bb.load(Bv, ks, lane);
      int u0 = 4 * (ks - 16) + kg;
      const __bf16* h0 = hr + 64 + 3 * u0;
      float p0 = bf(h0[0]), p1 = bf(h0[1]), p2 = bf(h0[2]);
      float q0 = bf(h0[6]), q1 = bf(h0[7]), q2 = bf(h0[8]);
      v16bf a;
#pragma unroll
      for (int q = 0; q < 8; ++q) {
        a[q]     = (__bf16)(p0 * bf(ev.s[3*q]) + p1 * bf(ev.s[3*q+1]) + p2 * bf(ev.s[3*q+2]));
        a[q + 8] = (__bf16)(q0 * bf(ev.s[3*q]) + q1 * bf(ev.s[3*q+1]) + q2 * bf(ev.s[3*q+2]));
      }
      bb.fire(a, acc);
    }
#pragma unroll
    for (int t = 0; t < 6; ++t)
#pragma unroll
      for (int rr = 0; rr < 8; ++rr)
        T[(rr + 8 * kg) * 192 + t * 16 + row] = acc[t][rr];
  }
  s2l1<0>(Bp + WS_S2L1, hr, er, T, lane, row, kg);
  s2l1<1>(Bp + WS_S2L1, hr, er, T, lane, row, kg);
  s2l1<2>(Bp + WS_S2L1, hr, er, T, lane, row, kg);
  __syncthreads();
  apply_gate(T, H, lane);
  __syncthreads();

  // ---- Stage 3: H x H -> T (dominant) ----
  { v8f acc[6] = {};
    hh_l0<6>(Bp + WS_S3L0, hr, acc, lane, kg);
#pragma unroll
    for (int t = 0; t < 6; ++t)
#pragma unroll
      for (int rr = 0; rr < 8; ++rr)
        T[(rr + 8 * kg) * 192 + t * 16 + row] = acc[t][rr];
  }
  s3l1<0>(Bp + WS_S3L1, hr, T, lane, row, kg);
  s3l1<1>(Bp + WS_S3L1, hr, T, lane, row, kg);
  s3l1<2>(Bp + WS_S3L1, hr, T, lane, row, kg);
  __syncthreads();
  apply_gate(T, H, lane);
  __syncthreads();

  // ---- Stage 4: H x H -> scalar ----
  { v8f acc[1] = {};
    hh_l0<1>(Bp + WS_S4, hr, acc, lane, kg);
    if (row == 0) {
#pragma unroll
      for (int rr = 0; rr < 8; ++rr)
        out[base + rr + 8 * kg] = acc[0][rr];
    }
  }
}

// ---------------- host ----------------
extern "C" void kernel_launch(void* const* d_in, const int* in_sizes, int n_in,
                              void* d_out, int out_size, void* d_ws, size_t ws_size,
                              hipStream_t stream)
{
  const float* s1 = (const float*)d_in[0];
  const float* s2 = (const float*)d_in[1];
  const float* ed = (const float*)d_in[2];
  const float* w1 = (const float*)d_in[3];
  const float* w2 = (const float*)d_in[4];
  const float* w3 = (const float*)d_in[5];
  const float* w4 = (const float*)d_in[6];
  __bf16* Bp = (__bf16*)d_ws;

  hipMemsetAsync(d_ws, 0, (size_t)WS_TOTAL * sizeof(__bf16), stream);

  const float r3 = 0.57735026919f, r6 = 0.40824829046f;
  const float a10 = sqrtf(1.f/1280.f), a11 = sqrtf(3.f/1280.f);
  const float a20 = sqrtf(1.f/768.f),  a22 = sqrtf(3.f/1024.f);
  const float a30 = sqrtf(1.f/5120.f), a32 = sqrtf(3.f/5120.f);
  const float a4  = sqrtf(1.f/5120.f);

  auto P = [&](const float* src, unsigned reg, int Klen, int Koff, int m2, int m3,
               int ncols, int coff, int Mpad, float sc) {
    int tot = Klen * ncols;
    prep_w<<<(tot + 255) / 256, 256, 0, stream>>>(src, Bp + reg, tot, ncols, Koff,
                                                  m2, m3, coff, Mpad, sc);
  };
  // stage1: (SITE,SITE)->H
  P(w1 + 0,      WS_S1L0, 1024,    0, 32, 64, 64,  0, 64, a10);       // (0,0,0)
  P(w1 + 98304,  WS_S1L0,  256, 1024, 16, 64, 64,  0, 64, a10 * r3);  // (1,1,0)
  P(w1 + 65536,  WS_S1L1,  512,    0, 16, 32, 32,  0, 32, a11 * r3);  // (0,1,1)
  P(w1 + 81920,  WS_S1L1,  512,  512, 32, 32, 32,  0, 32, a11 * r3);  // (1,0,1)
  P(w1 + 114688, WS_S1L1,  256, 1024, 16, 32, 32,  0, 32, a11 * r6);  // (1,1,1)
  // stage2: (H,EDGE)->GATED
  P(w2 + 0,      WS_S2L0,  512,    0,  8, 32, 32,  0, 96, a20);       // (0,0,0)
  P(w2 + 16384,  WS_S2L0,  512,    0,  8, 64, 64, 32, 96, a20);       // (0,0,1)
  P(w2 + 73728,  WS_S2L0,  256,  512,  8, 32, 32,  0, 96, a20 * r3);  // (1,1,0)
  P(w2 + 81920,  WS_S2L0,  256,  512,  8, 64, 64, 32, 96, a20 * r3);  // (1,1,1)
  P(w2 + 49152,  WS_S2L1,  512,    0,  8, 32, 32,  0, 32, a22 * r3);  // (0,1,2)
  P(w2 + 65536,  WS_S2L1,  256,  512,  8, 32, 32,  0, 32, a22 * r3);  // (1,0,2)
  P(w2 + 98304,  WS_S2L1,  256,  768,  8, 32, 32,  0, 32, a22 * r6);  // (1,1,2)
  // stage3: (H,H)->GATED
  P(w3 + 0,      WS_S3L0, 4096,    0, 64, 32, 32,  0, 96, a30);       // (0,0,0)
  P(w3 + 131072, WS_S3L0, 4096,    0, 64, 64, 64, 32, 96, a30);       // (0,0,1)
  P(w3 + 524288, WS_S3L0, 1024, 4096, 32, 32, 32,  0, 96, a30 * r3);  // (1,1,0)
  P(w3 + 557056, WS_S3L0, 1024, 4096, 32, 64, 64, 32, 96, a30 * r3);  // (1,1,1)
  P(w3 + 393216, WS_S3L1, 2048,    0, 32, 32, 32,  0, 32, a32 * r3);  // (0,1,2)
  P(w3 + 458752, WS_S3L1, 2048, 2048, 64, 32, 32,  0, 32, a32 * r3);  // (1,0,2)
  P(w3 + 622592, WS_S3L1, 1024, 4096, 32, 32, 32,  0, 32, a32 * r6);  // (1,1,2)
  // stage4: (H,H)->scalar
  P(w4 + 0,      WS_S4,   4096,    0, 64,  1,  1,  0, 16, a4);        // (0,0,0)
  P(w4 + 4096,   WS_S4,   1024, 4096, 32,  1,  1,  0, 16, a4 * r3);   // (1,1,0)

  int n = in_sizes[0] / 80;                 // 200000
  e3nn_fused<<<n / 32, 64, 0, stream>>>(s1, s2, ed, Bp, (float*)d_out);
}
